// GAT_85925115724064
// MI455X (gfx1250) — matrix operand order repaired and checked by reading, not compiled, not run on hardware
//
#include <hip/hip_runtime.h>
#include <hip/hip_fp16.h>
#include <math.h>

// GAT (2-layer, H=2, D=128) for gfx1250 (MI455X).
//  - GEMM h = x@W via v_wmma_f32_16x16x32_f16 (f16 in, f32 accum)
//    * W pre-transposed to N-major so both A and B fragments are
//      contiguous 16B global loads (no scalar u16 assembly)
//    * 4 N-tiles per wave: A fragment reused across 4 WMMAs per k-step
//  - edge softmax via L2-resident float atomics
//  - message scatter: one wave32 per edge, global_atomic_add_f32
//  - finalize: head mean + bias + L2 normalize, wave32 shuffle reduction

#define DFEAT 128
#define HHEADS 2
#define HD 256            // HHEADS * DFEAT
#define NEG_SLOPE 0.2f

typedef __attribute__((ext_vector_type(16))) _Float16 v16h;
typedef __attribute__((ext_vector_type(4)))  _Float16 v4h;
typedef __attribute__((ext_vector_type(8)))  float    v8f;

__device__ __forceinline__ float waveSum(float v) {
#pragma unroll
  for (int off = 16; off > 0; off >>= 1) v += __shfl_xor(v, off, 32);
  return v;
}

// float atomic max via ordered-integer encoding (works with -inf init)
__device__ __forceinline__ void atomicMaxF32(float* addr, float v) {
  if (v >= 0.0f) atomicMax((int*)addr, __float_as_int(v));
  else           atomicMin((unsigned int*)addr, __float_as_uint(v));
}

// ---------------- fp32 -> fp16 conversion (vectorized x4) ----------------
__global__ void gat_cvt_f16(const float* __restrict__ s, _Float16* __restrict__ d, int n4) {
  int i = blockIdx.x * blockDim.x + threadIdx.x;
  if (i >= n4) return;
  float4 v = *(const float4*)(s + (size_t)i * 4);
  v4h h;
  h.x = (_Float16)v.x; h.y = (_Float16)v.y; h.z = (_Float16)v.z; h.w = (_Float16)v.w;
  *(v4h*)(d + (size_t)i * 4) = h;
}

// ------------- W convert + transpose: [128,256] f32 -> [256,128] f16 -------------
__global__ void gat_cvt_wT(const float* __restrict__ W, _Float16* __restrict__ WT) {
  int i = blockIdx.x * blockDim.x + threadIdx.x;    // over 32768
  if (i >= DFEAT * HD) return;
  int k = i / HD, n = i % HD;
  WT[(size_t)n * DFEAT + k] = (_Float16)W[i];
}

// ---------------- GEMM: Hh[N,256] = X16[N,128] @ W (WT is [256,128]) ----------------
// One wave per 16x64 output strip (4 tiles); K=128 as 4 steps of
// v_wmma_f32_16x16x32_f16, A fragment reused across the 4 N-tiles.
__global__ void gat_gemm_wmma(const _Float16* __restrict__ X,
                              const _Float16* __restrict__ WT,
                              float* __restrict__ Hh, int nrows) {
  const int gw   = (blockIdx.x * blockDim.x + threadIdx.x) >> 5;
  const int lane = threadIdx.x & 31;
  const int groups_n = HD / 64;                 // 4 strips of 4 tiles
  const int tm = gw / groups_n;
  const int tg = gw % groups_n;                 // cols tg*64 .. tg*64+63
  if (tm * 16 >= nrows) return;                 // uniform per wave (EXEC all-1)
  const int m    = lane & 15;
  const int half = lane >> 4;
  int row = tm * 16 + m;
  if (row >= nrows) row = nrows - 1;            // clamp loads (stores guarded)

  v8f acc[4] = {{}, {}, {}, {}};
#pragma unroll
  for (int k0 = 0; k0 < DFEAT; k0 += 32) {
    // A 16x32 fragment: element j -> K = k0 + (j>>3)*16 + half*8 + (j&7)
    // => two contiguous 16B loads at K offsets half*8 and half*8+16
    const _Float16* ap = X + (size_t)row * DFEAT + k0 + half * 8;
    v16h a;
    *(float4*)&a         = *(const float4*)(ap);
    *(((float4*)&a) + 1) = *(const float4*)(ap + 16);
#pragma unroll
    for (int t = 0; t < 4; ++t) {
      // B 32x16 fragment: lane holds col n = tg*64 + t*16 + m;
      // elements j=0..15 -> K = k0 + (j>>3)*16 + half*8 + (j&7)
      const _Float16* bp =
          WT + (size_t)(tg * 64 + t * 16 + m) * DFEAT + k0 + half * 8;
      v16h b;
      *(float4*)&b         = *(const float4*)(bp);
      *(((float4*)&b) + 1) = *(const float4*)(bp + 16);
      acc[t] = __builtin_amdgcn_wmma_f32_16x16x32_f16(false, a, false, b,
                                                      (short)0, acc[t],
                                                      false, false);
    }
  }
  // C/D layout: VGPR r -> row half*8 + r, col = lane&15
#pragma unroll
  for (int t = 0; t < 4; ++t) {
#pragma unroll
    for (int r = 0; r < 8; ++r) {
      int orow = tm * 16 + half * 8 + r;
      if (orow < nrows)
        Hh[(size_t)orow * HD + tg * 64 + t * 16 + m] = acc[t][r];
    }
  }
}

// ---------------- per-node attention coefficients ----------------
// one wave per (node, head): alpha_s = <h, a_src>, alpha_d = <h, a_dst>
__global__ void gat_alpha(const float* __restrict__ Hh,
                          const float* __restrict__ a_src,
                          const float* __restrict__ a_dst,
                          float* __restrict__ asb, float* __restrict__ adb, int N) {
  const int gw   = (blockIdx.x * blockDim.x + threadIdx.x) >> 5;
  const int lane = threadIdx.x & 31;
  const int node = gw >> 1, h = gw & 1;
  if (node >= N) return;
  const int c = h * DFEAT + lane * 4;
  float4 hv = *(const float4*)(Hh + (size_t)node * HD + c);
  float4 sv = *(const float4*)(a_src + c);
  float4 dv = *(const float4*)(a_dst + c);
  float ss = hv.x * sv.x + hv.y * sv.y + hv.z * sv.z + hv.w * sv.w;
  float sd = hv.x * dv.x + hv.y * dv.y + hv.z * dv.z + hv.w * dv.w;
  ss = waveSum(ss);
  sd = waveSum(sd);
  if (lane == 0) { asb[node * 2 + h] = ss; adb[node * 2 + h] = sd; }
}

// ---------------- init accumulators ----------------
__global__ void gat_init(float* __restrict__ acc, float* __restrict__ m,
                         float* __restrict__ z, int N) {
  int i = blockIdx.x * blockDim.x + threadIdx.x;
  if (i < N * HD) acc[i] = 0.0f;
  if (i < N * HHEADS) { m[i] = -__builtin_inff(); z[i] = 0.0f; }
}

__device__ __forceinline__ void edge_ends(int eid, const int* __restrict__ ei,
                                          int E, int& s, int& d) {
  if (eid < E) { s = ei[eid]; d = ei[E + eid]; }
  else         { s = d = eid - E; }              // self loops appended
}

// ---------------- edge pass 1: leaky-relu logits + segment max ----------------
__global__ void gat_edge_logits(const int* __restrict__ ei, int E, int N,
                                const float* __restrict__ asb,
                                const float* __restrict__ adb,
                                float* __restrict__ ebuf, float* __restrict__ mbuf) {
  int eid = blockIdx.x * blockDim.x + threadIdx.x;
  if (eid >= E + N) return;
  int s, d; edge_ends(eid, ei, E, s, d);
#pragma unroll
  for (int h = 0; h < HHEADS; ++h) {
    float v = asb[s * 2 + h] + adb[d * 2 + h];
    v = v > 0.0f ? v : NEG_SLOPE * v;
    ebuf[(size_t)eid * 2 + h] = v;
    atomicMaxF32(&mbuf[d * 2 + h], v);
  }
}

// ---------------- edge pass 2: exp(e - max) + segment sum ----------------
__global__ void gat_edge_exp(const int* __restrict__ ei, int E, int N,
                             const float* __restrict__ mbuf,
                             float* __restrict__ ebuf, float* __restrict__ zbuf) {
  int eid = blockIdx.x * blockDim.x + threadIdx.x;
  if (eid >= E + N) return;
  int s, d; edge_ends(eid, ei, E, s, d);
#pragma unroll
  for (int h = 0; h < HHEADS; ++h) {
    float w = __expf(ebuf[(size_t)eid * 2 + h] - mbuf[d * 2 + h]);
    ebuf[(size_t)eid * 2 + h] = w;
    unsafeAtomicAdd(&zbuf[d * 2 + h], w);
  }
}

// ---------------- edge pass 3: scatter h[src]*alpha into acc[dst] ----------------
// one wave per edge; lane covers 4 channels per head; L2-resident atomics
__global__ void gat_scatter(const int* __restrict__ ei, int E, int N,
                            const float* __restrict__ Hh,
                            const float* __restrict__ ebuf,
                            const float* __restrict__ zbuf,
                            float* __restrict__ acc) {
  const int gw   = (blockIdx.x * blockDim.x + threadIdx.x) >> 5;
  const int lane = threadIdx.x & 31;
  if (gw >= E + N) return;
  int s, d; edge_ends(gw, ei, E, s, d);
  float a0 = ebuf[(size_t)gw * 2]     / (zbuf[d * 2]     + 1e-16f);
  float a1 = ebuf[(size_t)gw * 2 + 1] / (zbuf[d * 2 + 1] + 1e-16f);
  const float* hrow = Hh + (size_t)s * HD;
  float* orow = acc + (size_t)d * HD;
  const int c = lane * 4;
  float4 h0 = *(const float4*)(hrow + c);
  float4 h1 = *(const float4*)(hrow + DFEAT + c);
  unsafeAtomicAdd(&orow[c + 0], h0.x * a0);
  unsafeAtomicAdd(&orow[c + 1], h0.y * a0);
  unsafeAtomicAdd(&orow[c + 2], h0.z * a0);
  unsafeAtomicAdd(&orow[c + 3], h0.w * a0);
  unsafeAtomicAdd(&orow[DFEAT + c + 0], h1.x * a1);
  unsafeAtomicAdd(&orow[DFEAT + c + 1], h1.y * a1);
  unsafeAtomicAdd(&orow[DFEAT + c + 2], h1.z * a1);
  unsafeAtomicAdd(&orow[DFEAT + c + 3], h1.w * a1);
}

// ---------------- head mean + bias + L2 normalize ----------------
// one wave per node; writes f16 input for next layer, f32 to d_out on last layer
__global__ void gat_finalize(const float* __restrict__ acc,
                             const float* __restrict__ bias,
                             _Float16* __restrict__ x16,
                             float* __restrict__ xout, int N, int last) {
  const int gw   = (blockIdx.x * blockDim.x + threadIdx.x) >> 5;
  const int lane = threadIdx.x & 31;
  if (gw >= N) return;
  const int c = lane * 4;
  const float* r0 = acc + (size_t)gw * HD;
  float4 v0 = *(const float4*)(r0 + c);
  float4 v1 = *(const float4*)(r0 + DFEAT + c);
  float4 bv = *(const float4*)(bias + c);
  float4 v;
  v.x = 0.5f * (v0.x + v1.x) + bv.x;
  v.y = 0.5f * (v0.y + v1.y) + bv.y;
  v.z = 0.5f * (v0.z + v1.z) + bv.z;
  v.w = 0.5f * (v0.w + v1.w) + bv.w;
  float ss = waveSum(v.x * v.x + v.y * v.y + v.z * v.z + v.w * v.w);
  float inv = 1.0f / fmaxf(sqrtf(ss), 1e-12f);
  v.x *= inv; v.y *= inv; v.z *= inv; v.w *= inv;
  v4h h;
  h.x = (_Float16)v.x; h.y = (_Float16)v.y; h.z = (_Float16)v.z; h.w = (_Float16)v.w;
  *(v4h*)(x16 + (size_t)gw * DFEAT + c) = h;
  if (last) *(float4*)(xout + (size_t)gw * DFEAT + c) = v;
}

extern "C" void kernel_launch(void* const* d_in, const int* in_sizes, int n_in,
                              void* d_out, int out_size, void* d_ws, size_t ws_size,
                              hipStream_t stream) {
  const float* x     = (const float*)d_in[0];
  const int*   ei    = (const int*)d_in[1];     // [2,E] flat: row0=src, row1=dst
  const float* W     = (const float*)d_in[2];   // [L,128,256]
  const float* a_src = (const float*)d_in[3];   // [L,2,128]
  const float* a_dst = (const float*)d_in[4];
  const float* bias  = (const float*)d_in[5];   // [L,128]

  const int N    = in_sizes[0] / DFEAT;
  const int E    = in_sizes[1] / 2;
  const int Etot = E + N;

  // workspace carve (256B aligned)
  char* p = (char*)d_ws;
  auto take = [&](size_t bytes) -> void* {
    char* r = p; p += (bytes + 255) & ~(size_t)255; return (void*)r;
  };
  _Float16* x16 = (_Float16*)take((size_t)N * DFEAT * sizeof(_Float16));
  _Float16* WT  = (_Float16*)take((size_t)DFEAT * HD * sizeof(_Float16));
  float* Hh   = (float*)take((size_t)N * HD * sizeof(float));
  float* asb  = (float*)take((size_t)N * HHEADS * sizeof(float));
  float* adb  = (float*)take((size_t)N * HHEADS * sizeof(float));
  float* ebuf = (float*)take((size_t)Etot * HHEADS * sizeof(float));
  float* mbuf = (float*)take((size_t)N * HHEADS * sizeof(float));
  float* zbuf = (float*)take((size_t)N * HHEADS * sizeof(float));
  float* accb = (float*)take((size_t)N * HD * sizeof(float));

  auto cdiv = [](long a, long b) { return (int)((a + b - 1) / b); };
  const int BT = 256;

  gat_cvt_f16<<<cdiv((long)N * DFEAT / 4, BT), BT, 0, stream>>>(x, x16, N * DFEAT / 4);

  for (int l = 0; l < 2; ++l) {
    gat_cvt_wT<<<cdiv((long)DFEAT * HD, BT), BT, 0, stream>>>(
        W + (size_t)l * DFEAT * HD, WT);
    const int tilesM = cdiv(N, 16);
    gat_gemm_wmma<<<cdiv((long)tilesM * 4 * 32, BT), BT, 0, stream>>>(x16, WT, Hh, N);
    gat_alpha<<<cdiv((long)N * 2 * 32, BT), BT, 0, stream>>>(
        Hh, a_src + (size_t)l * HHEADS * DFEAT, a_dst + (size_t)l * HHEADS * DFEAT,
        asb, adb, N);
    gat_init<<<cdiv((long)N * HD, BT), BT, 0, stream>>>(accb, mbuf, zbuf, N);
    gat_edge_logits<<<cdiv(Etot, BT), BT, 0, stream>>>(ei, E, N, asb, adb, ebuf, mbuf);
    gat_edge_exp<<<cdiv(Etot, BT), BT, 0, stream>>>(ei, E, N, mbuf, ebuf, zbuf);
    gat_scatter<<<cdiv((long)Etot * 32, BT), BT, 0, stream>>>(ei, E, N, Hh, ebuf, zbuf, accb);
    gat_finalize<<<cdiv((long)N * 32, BT), BT, 0, stream>>>(
        accb, bias + (size_t)l * DFEAT, x16, (float*)d_out, N, l == 1 ? 1 : 0);
  }
}
